// LogisticRegression_50148038148444
// MI455X (gfx1250) — matrix operand order repaired and checked by reading, not compiled, not run on hardware
//
#include <hip/hip_runtime.h>
#include <hip/hip_bf16.h>
#include <math.h>

// ---------------- problem constants ----------------
#define NB    16          // batch
#define HL    200         // history length
#define EMBD  64
#define HID1  128
#define HID2  64
#define NSK   2000
#define PP    19900       // H*(H-1)/2 pairs per batch row
#define TPN   1244        // ceil(PP/16) 16-row tiles per batch row (even)
#define GPN   (TPN/2)     // 622 tile-pair groups per batch row
#define NGRP  (NB*GPN)    // 9952 wave-groups
#define ROWS  (NB*HL)     // 3200
#define XK    134         // 2*EMB + 6 time cats
#define KC1   5           // ceil(160/32) K-chunks layer1
#define NT1   8           // 128/16 N-tiles layer1
#define KC2   4           // 128/32
#define NT2   4           // 64/16

typedef __attribute__((ext_vector_type(16))) __bf16 v16bf;
typedef __attribute__((ext_vector_type(8)))  float  v8f;

union Frag { v16bf v; uint4 u[2]; unsigned short s[16]; };

__device__ __forceinline__ unsigned short f2bf(float f) {
  unsigned x = __float_as_uint(f);
  x += 0x7fffu + ((x >> 16) & 1u);          // round-to-nearest-even
  return (unsigned short)(x >> 16);
}
__device__ __forceinline__ float bf2f(unsigned short b) {
  return __uint_as_float(((unsigned)b) << 16);
}
__device__ __forceinline__ unsigned pk2(float a, float b) {
  return (unsigned)f2bf(a) | ((unsigned)f2bf(b) << 16);
}
__device__ __forceinline__ v8f wmma_bf16(const Frag& A, const Frag& B, v8f C) {
  return __builtin_amdgcn_wmma_f32_16x16x32_bf16(false, A.v, false, B.v,
                                                 (short)0, C, false, false);
}
// relu(acc+bias) -> bf16, packed 16B store of rows 8*khalf..8*khalf+7 of column col
__device__ __forceinline__ void store_relu_bf16(unsigned short* h, int col, int khalf,
                                                v8f acc, float bias) {
  float r0 = fmaxf(acc[0] + bias, 0.f), r1 = fmaxf(acc[1] + bias, 0.f);
  float r2 = fmaxf(acc[2] + bias, 0.f), r3 = fmaxf(acc[3] + bias, 0.f);
  float r4 = fmaxf(acc[4] + bias, 0.f), r5 = fmaxf(acc[5] + bias, 0.f);
  float r6 = fmaxf(acc[6] + bias, 0.f), r7 = fmaxf(acc[7] + bias, 0.f);
  uint4 pkd;
  pkd.x = pk2(r0, r1); pkd.y = pk2(r2, r3); pkd.z = pk2(r4, r5); pkd.w = pk2(r6, r7);
  *(uint4*)&h[col * 16 + 8 * khalf] = pkd;
}

// ---------------- setup kernels ----------------

// pair q -> (i, j), j < i, row-major over strictly-lower triangle
__global__ void k_pairs(unsigned* pairIJ) {
  int q = blockIdx.x * 256 + threadIdx.x;
  if (q >= PP) return;
  int i = (int)((1.0f + sqrtf(8.0f * (float)q + 1.0f)) * 0.5f);
  while (i * (i - 1) / 2 > q) --i;
  while ((i + 1) * i / 2 <= q) ++i;
  int j = q - i * (i - 1) / 2;
  pairIJ[q] = ((unsigned)i << 16) | (unsigned)j;
}

// gather embeddings per (n,h) -> bf16 table [ROWS][64]
__global__ void k_emb(const float* __restrict__ emb_table,
                      const int* __restrict__ skills,
                      unsigned short* __restrict__ embBF) {
  int idx = blockIdx.x * 256 + threadIdx.x;
  if (idx >= ROWS * EMBD) return;
  int row = idx >> 6, e = idx & 63;
  int s = skills[row];
  embBF[idx] = f2bf(emb_table[s * EMBD + e]);
}

// pre-swizzle weight matrix W[nOut][kIn] (row-major) into WMMA B-fragments:
// frag[((nt*kchunks+kc)*32 + lane)*16 + e] = W[nt*16+(lane&15)][kc*32 + (e&7) + 8*(lane>>4) + 16*(e>>3)]
__global__ void k_wfrag(const float* __restrict__ W, unsigned short* __restrict__ frag,
                        int nOut, int kIn, int ntiles, int kchunks) {
  int idx = blockIdx.x * 256 + threadIdx.x;
  int total = ntiles * kchunks * 32 * 16;
  if (idx >= total) return;
  int e    = idx & 15;
  int lane = (idx >> 4) & 31;
  int t    = idx >> 9;             // nt*kchunks + kc
  int kc   = t % kchunks;
  int nt   = t / kchunks;
  int ncol = nt * 16 + (lane & 15);
  int k    = kc * 32 + (e & 7) + 8 * (lane >> 4) + 16 * (e >> 3);
  float v = (ncol < nOut && k < kIn) ? W[ncol * kIn + k] : 0.0f;
  frag[idx] = f2bf(v);
}

__global__ void k_zero(float* p, int n) {
  int i = blockIdx.x * 256 + threadIdx.x;
  if (i < n) p[i] = 0.0f;
}

// ---------------- main pair-MLP kernel (one wave = two 16-pair tiles) ----------------
__global__ void __launch_bounds__(128) k_pairmlp(
    const unsigned* __restrict__ pairIJ,
    const unsigned short* __restrict__ embBF,
    const unsigned short* __restrict__ w1f,
    const unsigned short* __restrict__ w2f,
    const float* __restrict__ timestamps,
    const int* __restrict__ skills,
    const float* __restrict__ b1,
    const float* __restrict__ b2,
    const float* __restrict__ W3,
    const float* __restrict__ b3,
    float* __restrict__ v_sum) {
  // per-wave h1 tiles, column-major [col][m] so layer-1 stores are packed b128
  __shared__ __align__(16) unsigned short h1s[4][2][HID1 * 16];  // 32 KB

  const int lane  = threadIdx.x & 31;
  const int wave  = threadIdx.x >> 5;
  const int grp   = blockIdx.x * 4 + wave;          // grid sized exactly to NGRP/4
  const int n     = grp / GPN;
  const int tg    = grp - n * GPN;
  const int m     = lane & 15;
  const int khalf = lane >> 4;

  // ---- per-tile metadata + A-fragments for layer 1 (ISA 16-bit A 16x32 layout) ----
  // lane half 0: K = {0..7, 16..23}; lane half 1: K = {8..15, 24..31}
  Frag a[2][KC1];
  int  rowi[2];
  bool val[2];
#pragma unroll
  for (int tt = 0; tt < 2; ++tt) {
    int q = (tg * 2 + tt) * 16 + m;
    bool v = (q < PP);
    val[tt] = v;
    unsigned ij = pairIJ[v ? q : 0];
    int ii = (int)(ij >> 16), jj = (int)(ij & 0xffffu);
    int ri = n * HL + ii, rj = n * HL + jj;
    rowi[tt] = ri;
    float dt = timestamps[ri] - timestamps[rj];
    int cat = 1 + (dt > 1.0f) + (dt > 3600.0f) + (dt > 86400.0f) + (dt > 604800.0f);
    if (skills[ri] == 0 || skills[rj] == 0) cat = 0;
    const unsigned short* ei = embBF + ri * EMBD;
    const unsigned short* ej = embBF + rj * EMBD;
    int off = 8 * khalf;
    a[tt][0].u[0] = *(const uint4*)(ei + off);      a[tt][0].u[1] = *(const uint4*)(ei + off + 16);
    a[tt][1].u[0] = *(const uint4*)(ei + off + 32); a[tt][1].u[1] = *(const uint4*)(ei + off + 48);
    a[tt][2].u[0] = *(const uint4*)(ej + off);      a[tt][2].u[1] = *(const uint4*)(ej + off + 16);
    a[tt][3].u[0] = *(const uint4*)(ej + off + 32); a[tt][3].u[1] = *(const uint4*)(ej + off + 48);
#pragma unroll
    for (int e = 0; e < 16; ++e) {                  // one-hot time-cat chunk (x[128..159])
      int k = e + 8 * khalf + ((e >= 8) ? 8 : 0);
      a[tt][4].s[e] = (k == cat) ? (unsigned short)0x3F80u : (unsigned short)0u;
    }
  }

  // ---- layer 1: 8 N-tiles x 5 K-chunks; each B fragment feeds both M-tiles ----
  unsigned short* h10 = h1s[wave][0];
  unsigned short* h11 = h1s[wave][1];
#pragma unroll
  for (int nt = 0; nt < NT1; ++nt) {
    v8f acc0 = {}, acc1 = {};
#pragma unroll
    for (int kc = 0; kc < KC1; ++kc) {
      Frag b;
      const uint4* bp = (const uint4*)(w1f + (((nt * KC1 + kc) * 32 + lane) << 4));
      b.u[0] = bp[0]; b.u[1] = bp[1];
      acc0 = wmma_bf16(a[0][kc], b, acc0);
      acc1 = wmma_bf16(a[1][kc], b, acc1);
    }
    int col = nt * 16 + m;
    float bias = b1[col];
    store_relu_bf16(h10, col, khalf, acc0, bias);
    store_relu_bf16(h11, col, khalf, acc1, bias);
  }
  __syncthreads();

  // ---- layer 2 (re-swizzle h1 from LDS into A layout) with fused layer 3 ----
  Frag a2[2][KC2];
#pragma unroll
  for (int kc = 0; kc < KC2; ++kc) {
#pragma unroll
    for (int e = 0; e < 16; ++e) {
      int k = kc * 32 + e + 8 * khalf + ((e >= 8) ? 8 : 0);
      a2[0][kc].s[e] = h10[k * 16 + m];
      a2[1][kc].s[e] = h11[k * 16 + m];
    }
  }
  float p0[8] = {0,0,0,0,0,0,0,0};
  float p1[8] = {0,0,0,0,0,0,0,0};
#pragma unroll
  for (int nt = 0; nt < NT2; ++nt) {
    v8f acc0 = {}, acc1 = {};
#pragma unroll
    for (int kc = 0; kc < KC2; ++kc) {
      Frag b;
      const uint4* bp = (const uint4*)(w2f + (((nt * KC2 + kc) * 32 + lane) << 4));
      b.u[0] = bp[0]; b.u[1] = bp[1];
      acc0 = wmma_bf16(a2[0][kc], b, acc0);
      acc1 = wmma_bf16(a2[1][kc], b, acc1);
    }
    int col = nt * 16 + m;
    float bias = b2[col];
    float w3c  = W3[col];                 // W3 weight for this lane's h2 column
#pragma unroll
    for (int v = 0; v < 8; ++v) {
      p0[v] += fmaxf(acc0[v] + bias, 0.f) * w3c;
      p1[v] += fmaxf(acc1[v] + bias, 0.f) * w3c;
    }
  }

  // ---- layer 3 finish: butterfly all-reduce over the 16 columns in each lane-half.
  // After reduction every lane holds the 8 row-sums of its half (rows 8*khalf+v).
  float sel0 = 0.f, sel1 = 0.f;
#pragma unroll
  for (int v = 0; v < 8; ++v) {
    float x0 = p0[v], x1 = p1[v];
    x0 += __shfl_xor(x0, 1, 32);  x1 += __shfl_xor(x1, 1, 32);
    x0 += __shfl_xor(x0, 2, 32);  x1 += __shfl_xor(x1, 2, 32);
    x0 += __shfl_xor(x0, 4, 32);  x1 += __shfl_xor(x1, 4, 32);
    x0 += __shfl_xor(x0, 8, 32);  x1 += __shfl_xor(x1, 8, 32);
    if ((lane & 7) == v) { sel0 = x0; sel1 = x1; }
  }
  // lanes 0..7 own rows 0..7 (their own m), lanes 24..31 own rows 8..15 (their own m)
  if (lane < 8 || lane >= 24) {
    float bb = b3[0];
    if (val[0]) atomicAdd(&v_sum[rowi[0]], tanhf(sel0 + bb));
    if (val[1]) atomicAdd(&v_sum[rowi[1]], tanhf(sel1 + bb));
  }
}

// ---------------- final sparse linear + BCE / sigmoid / labels ----------------
__global__ void k_final(const float* __restrict__ users, const float* __restrict__ items,
                        const float* __restrict__ langs, const int* __restrict__ skills,
                        const float* __restrict__ targets, const unsigned char* __restrict__ mask,
                        const float* __restrict__ linW, const float* __restrict__ linb,
                        const float* __restrict__ v_sum, float* __restrict__ out) {
  int r = blockIdx.x * 256 + threadIdx.x;
  if (r >= ROWS) return;
  int n = r / HL;
  float acc = linb[0];
  const float* u = users + n * 100;
#pragma unroll 4
  for (int c = 0; c < 100; ++c) acc += u[c] * linW[c];
  const float* it = items + r * 100;
#pragma unroll 4
  for (int c = 0; c < 100; ++c) acc += it[c] * linW[100 + c];
  const float* lg = langs + r * 10;
#pragma unroll
  for (int c = 0; c < 10; ++c) acc += lg[c] * linW[200 + c];
  int s = skills[r];
  float vt  = (s != 0) ? v_sum[r] : 0.0f;      // padf
  float tgt = targets[r];
  float vc  = vt * tgt;                         // seg_sum(sim * targets[:,pi])
  acc += linW[210 + s] + linW[210 + NSK + s] * vt + linW[210 + 2 * NSK + s] * vc;
  float mk  = mask[r] ? 1.0f : 0.0f;
  float bce = fmaxf(acc, 0.0f) + log1pf(expf(-fabsf(acc))) - acc * tgt;  // logaddexp(0,x)-x*y
  out[r]            = bce * mk;
  out[ROWS + r]     = 1.0f / (1.0f + expf(-acc));
  out[2 * ROWS + r] = tgt;
}

// ---------------- launch ----------------
extern "C" void kernel_launch(void* const* d_in, const int* in_sizes, int n_in,
                              void* d_out, int out_size, void* d_ws, size_t ws_size,
                              hipStream_t stream) {
  const float* users      = (const float*)d_in[0];
  const float* items      = (const float*)d_in[1];
  const float* langs      = (const float*)d_in[2];
  const int*   skills     = (const int*)  d_in[3];
  const float* timestamps = (const float*)d_in[4];
  const float* targets    = (const float*)d_in[5];
  const unsigned char* mask = (const unsigned char*)d_in[6];
  const float* emb_table  = (const float*)d_in[7];
  const float* W1   = (const float*)d_in[8];
  const float* b1   = (const float*)d_in[9];
  const float* W2   = (const float*)d_in[10];
  const float* b2   = (const float*)d_in[11];
  const float* W3   = (const float*)d_in[12];
  const float* b3   = (const float*)d_in[13];
  const float* linW = (const float*)d_in[14];
  const float* linb = (const float*)d_in[15];
  float* out = (float*)d_out;

  // workspace layout (16B-aligned slices)
  char* ws = (char*)d_ws;
  unsigned*       pairIJ = (unsigned*)(ws + 0);                  // 79,600 B
  unsigned short* embBF  = (unsigned short*)(ws + 79616);        // 409,600 B
  unsigned short* w1f    = (unsigned short*)(ws + 489216);       // 40,960 B
  unsigned short* w2f    = (unsigned short*)(ws + 530176);       // 16,384 B
  float*          v_sum  = (float*)(ws + 546560);                // 12,800 B

  k_pairs<<<(PP + 255) / 256, 256, 0, stream>>>(pairIJ);
  k_emb<<<(ROWS * EMBD + 255) / 256, 256, 0, stream>>>(emb_table, skills, embBF);
  k_wfrag<<<(NT1 * KC1 * 32 * 16 + 255) / 256, 256, 0, stream>>>(W1, w1f, HID1, XK, NT1, KC1);
  k_wfrag<<<(NT2 * KC2 * 32 * 16 + 255) / 256, 256, 0, stream>>>(W2, w2f, HID2, HID1, NT2, KC2);
  k_zero<<<(ROWS + 255) / 256, 256, 0, stream>>>(v_sum, ROWS);

  k_pairmlp<<<NGRP / 4, 128, 0, stream>>>(pairIJ, embBF, w1f, w2f, timestamps, skills,
                                          b1, b2, W3, b3, v_sum);

  k_final<<<(ROWS + 255) / 256, 256, 0, stream>>>(users, items, langs, skills, targets, mask,
                                                  linW, linb, v_sum, out);
}
